// NNGraphNet_65343632441457
// MI455X (gfx1250) — matrix operand order repaired and checked by reading, not compiled, not run on hardware
//
#include <hip/hip_runtime.h>
#include <hip/hip_bf16.h>
#include <math.h>

#define NN 20000
#define EE 50000
#define FF 64
#define EPSV 1e-5f

typedef __attribute__((ext_vector_type(16))) __bf16 v16bf;
typedef __attribute__((ext_vector_type(8)))  float  v8f;

__device__ inline v16bf zero_v16bf() {
    v16bf z;
    #pragma unroll
    for (int p = 0; p < 16; p++) z[p] = (__bf16)0.0f;
    return z;
}

// ---------------------------------------------------------------------------
// weight swizzlers: put WMMA operands in fragment-native contiguous layout
// B-fragment element p of lane l for (tile j, kstep ks):
//   k = ks*32 + (p>>3)*16 + (l>>4)*8 + (p&7) ; c = j*16 + (l&15)
// stored at ((j*KS + ks)*32 + l)*16 + p  -> each lane reads 32 contiguous bytes
// ---------------------------------------------------------------------------
__global__ void w3_swizzle_kernel(const float* __restrict__ src, __bf16* __restrict__ dst) {
    int o = blockIdx.x * blockDim.x + threadIdx.x;       // 256 tiles * 4 ks * 32 lanes * 16
    if (o >= 256 * 4 * 32 * 16) return;
    int p    = o & 15;
    int lane = (o >> 4) & 31;
    int ks   = (o >> 9) & 3;
    int j    = o >> 11;
    int k = ks * 32 + ((p >> 3) << 4) + (lane >> 4) * 8 + (p & 7);
    int c = j * 16 + (lane & 15);
    dst[o] = (__bf16)src[k * 4096 + c];
}

__global__ void ln_swizzle_kernel(const float* __restrict__ src, __bf16* __restrict__ dst) {
    int o = blockIdx.x * blockDim.x + threadIdx.x;       // 4 tiles * 2 ks * 32 lanes * 16
    if (o >= 4 * 2 * 32 * 16) return;
    int p    = o & 15;
    int lane = (o >> 4) & 31;
    int ks   = (o >> 9) & 1;
    int gt   = o >> 10;
    int k = ks * 32 + ((p >> 3) << 4) + (lane >> 4) * 8 + (p & 7);
    int c = gt * 16 + (lane & 15);
    dst[o] = (__bf16)src[k * 64 + c];
}

// ---------------------------------------------------------------------------
// small helpers
// ---------------------------------------------------------------------------
__global__ void deg_kernel(const int* __restrict__ dst, float* __restrict__ deg, int e) {
    int i = blockIdx.x * blockDim.x + threadIdx.x;
    if (i < e) atomicAdd(&deg[dst[i]], 1.0f);
}

__global__ void finalize_gln_kernel(float* gstats, float count) {
    float mu  = gstats[0] / count;
    float var = gstats[1] / count - mu * mu;
    var = fmaxf(var, 0.0f);
    gstats[2] = mu;
    gstats[3] = 1.0f / (sqrtf(var) + EPSV);   // graph_ln: xc / (std + eps)
}

__global__ void apply_gln_kernel(const float* __restrict__ tmp, const float* __restrict__ gstats,
                                 const float* __restrict__ w, const float* __restrict__ b,
                                 float* __restrict__ h, int total) {
    int i = blockIdx.x * blockDim.x + threadIdx.x;
    if (i >= total) return;
    int f = i & (FF - 1);
    h[i] = (tmp[i] - gstats[2]) * gstats[3] * w[f] + b[f];
}

__global__ void apply_gln_residual_kernel(const float* __restrict__ tmp, const float* __restrict__ gstats,
                                          const float* __restrict__ w, const float* __restrict__ b,
                                          float* __restrict__ h, int total) {
    int i = blockIdx.x * blockDim.x + threadIdx.x;
    if (i >= total) return;
    int f = i & (FF - 1);
    h[i] = h[i] + (tmp[i] - gstats[2]) * gstats[3] * w[f] + b[f];
}

// ---------------------------------------------------------------------------
// encoder MLP (5->64->64->64->64, relu x3) + global-LN stats accumulation
// ---------------------------------------------------------------------------
__global__ void enc_kernel(const float* __restrict__ x,
                           const float* W0, const float* b0, const float* W1, const float* b1,
                           const float* W2, const float* b2, const float* W3, const float* b3,
                           float* __restrict__ tmp, float* __restrict__ gstats, int n) {
    __shared__ float buf[4][FF];
    int w = threadIdx.x >> 5, lane = threadIdx.x & 31;
    int node = blockIdx.x * 4 + w;
    bool act = node < n;
    int o0 = lane, o1 = lane + 32;
    float a0 = 0.f, a1 = 0.f;
    if (act) {
        a0 = b0[o0]; a1 = b0[o1];
        #pragma unroll
        for (int i = 0; i < 5; i++) {
            float v = x[node * 5 + i];
            a0 += v * W0[i * FF + o0]; a1 += v * W0[i * FF + o1];
        }
        a0 = fmaxf(a0, 0.f); a1 = fmaxf(a1, 0.f);
        buf[w][o0] = a0; buf[w][o1] = a1;
    }
    __syncthreads();
    const float* Ws[3] = {W1, W2, W3};
    const float* bs[3] = {b1, b2, b3};
    for (int L = 0; L < 3; L++) {
        float c0 = 0.f, c1 = 0.f;
        if (act) {
            c0 = bs[L][o0]; c1 = bs[L][o1];
            #pragma unroll 8
            for (int i = 0; i < FF; i++) {
                float v = buf[w][i];
                c0 += v * Ws[L][i * FF + o0]; c1 += v * Ws[L][i * FF + o1];
            }
            if (L < 2) { c0 = fmaxf(c0, 0.f); c1 = fmaxf(c1, 0.f); }
        }
        __syncthreads();
        if (act) { buf[w][o0] = c0; buf[w][o1] = c1; }
        __syncthreads();
        a0 = c0; a1 = c1;
    }
    if (act) {
        tmp[node * FF + o0] = a0; tmp[node * FF + o1] = a1;
        float s = a0 + a1, q = a0 * a0 + a1 * a1;
        #pragma unroll
        for (int m = 16; m >= 1; m >>= 1) { s += __shfl_xor(s, m, 32); q += __shfl_xor(q, m, 32); }
        if (lane == 0) { atomicAdd(&gstats[0], s); atomicAdd(&gstats[1], q); }
    }
}

// ---------------------------------------------------------------------------
// edge hidden layers: 3->128 relu, 128->128 relu ; output bf16 in WMMA
// A-fragment-native order: e2swz[((edge*4 + ks)*2 + half)*16 + p]
// ---------------------------------------------------------------------------
__global__ void edge_hidden_kernel(const float* __restrict__ ea,
                                   const float* W1, const float* b1,
                                   const float* W2, const float* b2,
                                   __bf16* __restrict__ e2swz, int e) {
    __shared__ float e1[128];
    int t = threadIdx.x;
    int ed = blockIdx.x;
    if (ed >= e) return;
    float a = b1[t];
    #pragma unroll
    for (int i = 0; i < 3; i++) a += ea[ed * 3 + i] * W1[i * 128 + t];
    a = fmaxf(a, 0.f);
    e1[t] = a;
    __syncthreads();
    float c = b2[t];
    #pragma unroll 8
    for (int i = 0; i < 128; i++) c += e1[i] * W2[i * 128 + t];
    c = fmaxf(c, 0.f);
    // scatter into A-fragment layout: k = t
    int ks = t >> 5, kl = t & 31;
    int hf = (kl >> 3) & 1;
    int vr = ((kl >> 4) & 1) * 4 + ((kl >> 1) & 3);
    int p  = vr * 2 + (kl & 1);
    e2swz[((ed * 4 + ks) * 2 + hf) * 16 + p] = (__bf16)c;
}

// ---------------------------------------------------------------------------
// fused NNConv edge kernel: one 256-thread block (8 waves) owns 32 edges.
//  phase A: S2 = hs@LNw, S3 = hs@LNb (4 WMMAs, pre-swizzled B frags)
//  phase B: theta = e2@W3 + b3 via bf16 WMMA, 2-wide column tiles for ILP;
//           store theta*ln_w (bf16) into 256KB dynamic LDS; LN stats in regs
//  phase C: per-edge LN finalize; contraction msg = inv*S1 - mu*inv*S2 + S3;
//           f32 atomic scatter into agg
// ---------------------------------------------------------------------------
__global__ void nnconv_fused_kernel(const __bf16* __restrict__ e2swz,
                                    const __bf16* __restrict__ W3swz,
                                    const float* __restrict__ b3,
                                    const float* __restrict__ lnw_raw,
                                    const __bf16* __restrict__ lnw_swz,
                                    const __bf16* __restrict__ lnb_swz,
                                    const float* __restrict__ h,
                                    const int* __restrict__ srcI, const int* __restrict__ dstI,
                                    float* __restrict__ agg, int e) {
    extern __shared__ char smem[];
    __bf16* thetaW = (__bf16*)smem;                          // [2][64][16][64] bf16 = 256KB
    float*  hs     = (float*)(smem + 262144);                // [32][64] f32 = 8KB
    float*  S2     = (float*)(smem + 262144 + 8192);         // [32][64] f32 = 8KB
    float*  S3     = (float*)(smem + 262144 + 16384);        // [32][64] f32 = 8KB
    float*  ssum   = (float*)(smem + 262144 + 24576);        // 32
    float*  sqsum  = ssum + 32;
    float*  muS    = ssum + 64;
    float*  invS   = ssum + 96;
    int*    dsts   = (int*)(ssum + 128);                     // 32

    const int t = threadIdx.x;
    const int w = t >> 5, lane = t & 31;
    const int half = lane >> 4, nlane = lane & 15;
    const int tile = blockIdx.x * 32;

    // ---- stage hs (gathered h[src]) + dst + zero stats ----
    for (int idx = t; idx < 32 * FF; idx += 256) {
        int row = idx >> 6, col = idx & 63;
        int ed = tile + row;
        hs[idx] = (ed < e) ? h[srcI[ed] * FF + col] : 0.0f;
    }
    if (t < 32) {
        dsts[t]  = (tile + t < e) ? dstI[tile + t] : 0;
        ssum[t]  = 0.f;
        sqsum[t] = 0.f;
    }
    __syncthreads();

    // ---- phase A: S2/S3 via WMMA. wave w -> (sub2 = w>>2, gt = w&3) ----
    {
        int sub2 = w >> 2, gt = w & 3;
        int mrow = sub2 * 16 + nlane;
        v8f accW = {}, accB = {};
        #pragma unroll
        for (int ks2 = 0; ks2 < 2; ks2++) {
            v16bf Ah;
            #pragma unroll
            for (int p = 0; p < 16; p++) {
                int vr = p >> 1;
                int kloc = ((vr >> 2) << 4) + (half << 3) + ((vr & 3) << 1) + (p & 1);
                Ah[p] = (__bf16)hs[mrow * FF + ks2 * 32 + kloc];
            }
            v16bf Bw = *(const v16bf*)(lnw_swz + ((gt * 2 + ks2) * 32 + lane) * 16);
            v16bf Bb = *(const v16bf*)(lnb_swz + ((gt * 2 + ks2) * 32 + lane) * 16);
            accW = __builtin_amdgcn_wmma_f32_16x16x32_bf16(false, Ah, false, Bw, (short)0, accW, false, false);
            accB = __builtin_amdgcn_wmma_f32_16x16x32_bf16(false, Ah, false, Bb, (short)0, accB, false, false);
        }
        #pragma unroll
        for (int r = 0; r < 8; r++) {
            int er = sub2 * 16 + r + half * 8;
            S2[er * FF + gt * 16 + nlane] = accW[r];
            S3[er * FF + gt * 16 + nlane] = accB[r];
        }
    }

    // ---- phase B: theta GEMM. wave w -> (sub = w>>2, j-quarter q = w&3) ----
    {
        int sub = w >> 2, q = w & 3;
        // A fragments (contiguous 32B loads, pre-swizzled)
        v16bf A[4];
        int m = tile + sub * 16 + nlane;
        bool mok = m < e;
        #pragma unroll
        for (int ks = 0; ks < 4; ks++) {
            A[ks] = mok ? *(const v16bf*)(e2swz + ((m * 4 + ks) * 2 + half) * 16)
                        : zero_v16bf();
        }
        float s8[8], q8[8];
        #pragma unroll
        for (int r = 0; r < 8; r++) { s8[r] = 0.f; q8[r] = 0.f; }

        for (int jj = 0; jj < 32; jj++) {
            int j0 = q * 64 + 2 * jj;          // two independent column tiles
            int j1 = j0 + 1;
            v8f acc0 = {}, acc1 = {};
            #pragma unroll
            for (int ks = 0; ks < 4; ks++) {
                v16bf B0 = *(const v16bf*)(W3swz + ((j0 * 4 + ks) * 32 + lane) * 16);
                v16bf B1 = *(const v16bf*)(W3swz + ((j1 * 4 + ks) * 32 + lane) * 16);
                acc0 = __builtin_amdgcn_wmma_f32_16x16x32_bf16(false, A[ks], false, B0, (short)0, acc0, false, false);
                acc1 = __builtin_amdgcn_wmma_f32_16x16x32_bf16(false, A[ks], false, B1, (short)0, acc1, false, false);
            }
            int c0 = j0 * 16 + nlane, c1 = j1 * 16 + nlane;
            float bc0 = b3[c0], bc1 = b3[c1];
            float wc0 = lnw_raw[c0], wc1 = lnw_raw[c1];
            int f0 = j0 >> 2;                                  // same f for j0,j1
            int g0 = 16 * (j0 & 3) + nlane, g1 = 16 * (j1 & 3) + nlane;
            #pragma unroll
            for (int r = 0; r < 8; r++) {
                int er = r + half * 8;
                float t0 = acc0[r] + bc0;
                float t1 = acc1[r] + bc1;
                s8[r] += t0 + t1;
                q8[r] += t0 * t0 + t1 * t1;
                int base = sub * 65536 + f0 * 1024 + er * FF;  // [sub][f][e][g]
                thetaW[base + g0] = (__bf16)(t0 * wc0);
                thetaW[base + g1] = (__bf16)(t1 * wc1);
            }
        }
        // per-row partial stats: reduce across the 16 lanes of each half
        #pragma unroll
        for (int r = 0; r < 8; r++) {
            #pragma unroll
            for (int msk = 8; msk >= 1; msk >>= 1) {
                s8[r] += __shfl_xor(s8[r], msk, 32);
                q8[r] += __shfl_xor(q8[r], msk, 32);
            }
        }
        if (nlane == 0) {
            #pragma unroll
            for (int r = 0; r < 8; r++) {
                int er = sub * 16 + r + half * 8;
                atomicAdd(&ssum[er], s8[r]);
                atomicAdd(&sqsum[er], q8[r]);
            }
        }
    }
    __syncthreads();

    if (t < 32) {
        float mu  = ssum[t] * (1.0f / 4096.0f);
        float var = sqsum[t] * (1.0f / 4096.0f) - mu * mu;
        muS[t]  = mu;
        invS[t] = rsqrtf(fmaxf(var, 0.f) + EPSV);   // _ln: 1/sqrt(var+eps)
    }
    __syncthreads();

    // ---- phase C: contraction + scatter. thread t -> edge eg = t&31, g-range (t>>5)*8 ----
    {
        int eg = t & 31;
        int g0 = (t >> 5) * 8;
        float acc8[8];
        #pragma unroll
        for (int i = 0; i < 8; i++) acc8[i] = 0.f;
        const __bf16* tp = thetaW + (eg >> 4) * 65536 + (eg & 15) * FF + g0;
        #pragma unroll 4
        for (int f = 0; f < FF; f++) {
            float hsv = hs[eg * FF + f];
            uint4 uv = *(const uint4*)(tp + f * 1024);       // 8 bf16, 16B aligned
            acc8[0] += hsv * __uint_as_float((uv.x & 0xffffu) << 16);
            acc8[1] += hsv * __uint_as_float(uv.x & 0xffff0000u);
            acc8[2] += hsv * __uint_as_float((uv.y & 0xffffu) << 16);
            acc8[3] += hsv * __uint_as_float(uv.y & 0xffff0000u);
            acc8[4] += hsv * __uint_as_float((uv.z & 0xffffu) << 16);
            acc8[5] += hsv * __uint_as_float(uv.z & 0xffff0000u);
            acc8[6] += hsv * __uint_as_float((uv.w & 0xffffu) << 16);
            acc8[7] += hsv * __uint_as_float(uv.w & 0xffff0000u);
        }
        int ed = tile + eg;
        if (ed < e) {
            float mu = muS[eg], iv = invS[eg];
            int dn = dsts[eg];
            #pragma unroll
            for (int i = 0; i < 8; i++) {
                float msg = iv * acc8[i] - mu * iv * S2[eg * FF + g0 + i] + S3[eg * FF + g0 + i];
                atomicAdd(&agg[dn * FF + g0 + i], msg);
            }
        }
    }
}

// ---------------------------------------------------------------------------
// node update: out = agg/denom + h@root + bias ; relu ; LN-stats accumulation
// ---------------------------------------------------------------------------
__global__ void node_update_kernel(const float* __restrict__ h, const float* __restrict__ agg,
                                   const float* __restrict__ deg,
                                   const float* __restrict__ root, const float* __restrict__ bias,
                                   float* __restrict__ tmp, float* __restrict__ gstats, int n) {
    __shared__ float buf[4][FF];
    int w = threadIdx.x >> 5, lane = threadIdx.x & 31;
    int node = blockIdx.x * 4 + w;
    bool act = node < n;
    int o0 = lane, o1 = lane + 32;
    if (act) { buf[w][o0] = h[node * FF + o0]; buf[w][o1] = h[node * FF + o1]; }
    __syncthreads();
    if (!act) return;
    float dn = fmaxf(deg[node], 1.0f);
    float r0 = agg[node * FF + o0] / dn + bias[o0];
    float r1 = agg[node * FF + o1] / dn + bias[o1];
    #pragma unroll 8
    for (int i = 0; i < FF; i++) {
        float v = buf[w][i];
        r0 += v * root[i * FF + o0]; r1 += v * root[i * FF + o1];
    }
    r0 = fmaxf(r0, 0.f); r1 = fmaxf(r1, 0.f);
    tmp[node * FF + o0] = r0; tmp[node * FF + o1] = r1;
    float s = r0 + r1, q = r0 * r0 + r1 * r1;
    #pragma unroll
    for (int m = 16; m >= 1; m >>= 1) { s += __shfl_xor(s, m, 32); q += __shfl_xor(q, m, 32); }
    if (lane == 0) { atomicAdd(&gstats[0], s); atomicAdd(&gstats[1], q); }
}

// ---------------------------------------------------------------------------
// decoder MLP: 64->64->64->64 relu x3, 64->3 final
// ---------------------------------------------------------------------------
__global__ void dec_kernel(const float* __restrict__ h,
                           const float* W0, const float* b0, const float* W1, const float* b1,
                           const float* W2, const float* b2, const float* W3, const float* b3,
                           float* __restrict__ out, int n) {
    __shared__ float buf[4][FF];
    int w = threadIdx.x >> 5, lane = threadIdx.x & 31;
    int node = blockIdx.x * 4 + w;
    bool act = node < n;
    int o0 = lane, o1 = lane + 32;
    if (act) { buf[w][o0] = h[node * FF + o0]; buf[w][o1] = h[node * FF + o1]; }
    __syncthreads();
    const float* Ws[3] = {W0, W1, W2};
    const float* bs[3] = {b0, b1, b2};
    for (int L = 0; L < 3; L++) {
        float c0 = 0.f, c1 = 0.f;
        if (act) {
            c0 = bs[L][o0]; c1 = bs[L][o1];
            #pragma unroll 8
            for (int i = 0; i < FF; i++) {
                float v = buf[w][i];
                c0 += v * Ws[L][i * FF + o0]; c1 += v * Ws[L][i * FF + o1];
            }
            c0 = fmaxf(c0, 0.f); c1 = fmaxf(c1, 0.f);
        }
        __syncthreads();
        if (act) { buf[w][o0] = c0; buf[w][o1] = c1; }
        __syncthreads();
    }
    if (act && lane < 3) {
        float acc = b3[lane];
        #pragma unroll 8
        for (int i = 0; i < FF; i++) acc += buf[w][i] * W3[i * 3 + lane];
        out[node * 3 + lane] = acc;
    }
}

// ---------------------------------------------------------------------------
// host launcher
// ---------------------------------------------------------------------------
#define PF(i) ((const float*)d_in[(i)])

extern "C" void kernel_launch(void* const* d_in, const int* in_sizes, int n_in,
                              void* d_out, int out_size, void* d_ws, size_t ws_size,
                              hipStream_t stream) {
    (void)in_sizes; (void)n_in; (void)out_size; (void)ws_size;
    const float* x    = PF(0);
    const int*   ei   = (const int*)d_in[1];
    const int*   srcI = ei;
    const int*   dstI = ei + EE;
    const float* ea   = PF(2);
    // param leaf order (insertion order): enc 3..10 ; enc_ln 11,12 ;
    // layer l base = 13 + 12*l : eW1,eb1,eW2,eb2,eW3,eb3,elnw,elnb,root,bias,clnw,clnb ;
    // dec 37..44

    char* p = (char*)d_ws;
    auto carve = [&](size_t bytes) { void* r = (void*)p; p += (bytes + 255) & ~(size_t)255; return r; };
    float*  h       = (float*)carve((size_t)NN * FF * 4);
    float*  tmp     = (float*)carve((size_t)NN * FF * 4);
    float*  agg     = (float*)carve((size_t)NN * FF * 4);
    float*  deg     = (float*)carve((size_t)NN * 4);
    __bf16* e2swz   = (__bf16*)carve((size_t)EE * 128 * 2);
    __bf16* W3swz   = (__bf16*)carve((size_t)128 * 4096 * 2);
    __bf16* lnw_swz = (__bf16*)carve((size_t)64 * 64 * 2);
    __bf16* lnb_swz = (__bf16*)carve((size_t)64 * 64 * 2);
    float*  gst     = (float*)carve(4 * 4);

    const int fblocks = (EE + 31) / 32;                    // 32 edges / block
    const int FSMEM   = 262144 + 3 * 8192 + 512 + 128;     // 287,360 B dynamic LDS

    hipMemsetAsync(deg, 0, (size_t)NN * 4, stream);
    hipMemsetAsync(gst, 0, 16, stream);

    enc_kernel<<<(NN + 3) / 4, 128, 0, stream>>>(x, PF(3), PF(4), PF(5), PF(6), PF(7), PF(8),
                                                 PF(9), PF(10), tmp, gst, NN);
    finalize_gln_kernel<<<1, 1, 0, stream>>>(gst, (float)NN * FF);
    apply_gln_kernel<<<(NN * FF + 255) / 256, 256, 0, stream>>>(tmp, gst, PF(11), PF(12), h, NN * FF);

    deg_kernel<<<(EE + 255) / 256, 256, 0, stream>>>(dstI, deg, EE);

    for (int l = 0; l < 2; l++) {
        int base = 13 + 12 * l;
        w3_swizzle_kernel<<<(128 * 4096 + 255) / 256, 256, 0, stream>>>(PF(base + 4), W3swz);
        ln_swizzle_kernel<<<(64 * 64 + 255) / 256, 256, 0, stream>>>(PF(base + 6), lnw_swz);
        ln_swizzle_kernel<<<(64 * 64 + 255) / 256, 256, 0, stream>>>(PF(base + 7), lnb_swz);
        edge_hidden_kernel<<<EE, 128, 0, stream>>>(ea, PF(base + 0), PF(base + 1),
                                                   PF(base + 2), PF(base + 3), e2swz, EE);
        hipMemsetAsync(agg, 0, (size_t)NN * FF * 4, stream);
        nnconv_fused_kernel<<<fblocks, 256, FSMEM, stream>>>(e2swz, W3swz, PF(base + 5),
                                                             PF(base + 6), lnw_swz, lnb_swz,
                                                             h, srcI, dstI, agg, EE);
        hipMemsetAsync(gst, 0, 16, stream);
        node_update_kernel<<<(NN + 3) / 4, 128, 0, stream>>>(h, agg, deg, PF(base + 8), PF(base + 9),
                                                             tmp, gst, NN);
        finalize_gln_kernel<<<1, 1, 0, stream>>>(gst, (float)NN * FF);
        apply_gln_residual_kernel<<<(NN * FF + 255) / 256, 256, 0, stream>>>(tmp, gst, PF(base + 10),
                                                                             PF(base + 11), h, NN * FF);
    }

    dec_kernel<<<(NN + 3) / 4, 128, 0, stream>>>(h, PF(37), PF(38), PF(39), PF(40),
                                                 PF(41), PF(42), PF(43), PF(44),
                                                 (float*)d_out, NN);
}
